// Classifier_29171417874733
// MI455X (gfx1250) — matrix operand order repaired and checked by reading, not compiled, not run on hardware
//
#include <hip/hip_runtime.h>
#include <hip/hip_bf16.h>
#include <math.h>

typedef __bf16 bf16_t;
typedef __attribute__((ext_vector_type(16))) __bf16 v16bf;
typedef __attribute__((ext_vector_type(8)))  __bf16 v8bf;
typedef __attribute__((ext_vector_type(8)))  float  v8f;
typedef unsigned int v4u __attribute__((ext_vector_type(4)));
typedef int          v8i __attribute__((ext_vector_type(8)));
typedef int          v4i __attribute__((ext_vector_type(4)));

#define B_ROWS   8192
#define D_IN     2048
#define D_BOT    1024
#define NCLS     1000
#define NPAD     1024
#define SHARE_N  500
#define INV_TEMP 20.0f
#define NEG_FILL -1e9f
#define NORM_EPS 1e-12f

// LDS tile row: 32 bf16 = 64B data + 16B TDM pad (pad_interval=16dw, pad_amount=4dw)
// -> 80B stride: 16 consecutive rows hit distinct bank groups (conflict-free b128)
#define LDS_ROW_B 80

// ---------------------------------------------------------------------------
// Tensor Data Mover: 2D tile load Global -> LDS (bf16, padded rows).
// D# group0: count=1 | lds_addr | global_addr | type=2
// D# group1: data_size=2B, pad_enable, pad_interval=16dw, pad_amount=4dw,
//            tensor_dim0/1, tile_dim0/1, tensor_dim0_stride
// ---------------------------------------------------------------------------
static __device__ __forceinline__ void tdm_load_tile(unsigned lds_off,
                                                     const bf16_t* gptr,
                                                     int tensor_d0, int tensor_d1,
                                                     int tile_d0, int tile_d1,
                                                     int stride0) {
  unsigned long long ga = (unsigned long long)(uintptr_t)gptr;
  v4u g0;
  g0[0] = 1u;                                        // count=1, user descriptor
  g0[1] = lds_off;                                   // lds_addr
  g0[2] = (unsigned)(ga & 0xFFFFFFFFu);              // global_addr[31:0]
  g0[3] = (unsigned)((ga >> 32) & 0x01FFFFFFu)       // global_addr[56:32]
        | (2u << 30);                                // type = 2 ("image")
  v8i g1;
  g1[0] = (int)((1u << 16)                           // data_size = 2 bytes
              | (1u << 20)                           // pad_enable
              | (3u << 22)                           // pad_interval: 16 dwords
              | (3u << 25));                         // pad_amount: 4 dwords
  g1[1] = (int)(((unsigned)tensor_d0 & 0xFFFFu) << 16);
  g1[2] = (int)(((unsigned)tensor_d0 >> 16) | (((unsigned)tensor_d1 & 0xFFFFu) << 16));
  g1[3] = (int)(((unsigned)tensor_d1 >> 16) | ((unsigned)tile_d0 << 16));
  g1[4] = tile_d1;                                   // tile_dim1 (tile_dim2 = 0)
  g1[5] = stride0;                                   // tensor_dim0_stride[31:0]
  g1[6] = 0;
  g1[7] = 0;
  v4i gz = {0, 0, 0, 0};
#if __clang_major__ >= 23
  v8i gz8 = {0, 0, 0, 0, 0, 0, 0, 0};
  __builtin_amdgcn_tensor_load_to_lds(g0, g1, gz, gz, gz8, 0);
#else
  __builtin_amdgcn_tensor_load_to_lds(g0, g1, gz, gz, 0);
#endif
}

static __device__ __forceinline__ unsigned lds_offset_of(const void* p) {
  return (unsigned)(uintptr_t)p;   // LDS aperture: addr[31:0] = LDS offset
}

static __device__ __forceinline__ v16bf lds_frag(const unsigned char* p0,
                                                 const unsigned char* p1) {
  v8bf c0 = *(const v8bf*)p0;
  v8bf c1 = *(const v8bf*)p1;
  v16bf r;
#pragma unroll
  for (int i = 0; i < 8; ++i) { r[i] = c0[i]; r[i + 8] = c1[i]; }
  return r;
}

// ---------------------------------------------------------------------------
// C[M,N] = A[M,K] * B[N,K]^T  (bf16 via WMMA, f32 accumulate)
// WG 256 thr = 8 waves -> 256x128 tile; wave tile 64x64 (4x4 accs).
// A/B tiles staged in LDS by TDM, double-buffered, TENSORcnt-paced.
// MODE 0: += bias[n];  store f32 (ldc) + bf16 (NPAD)               [GEMM1]
// MODE 1: store f32 if n < NCLS                                    [GEMM2]
// MODE 2: += addend[m*NPAD+n]; store f32                           [GEMM3]
// MODE 3: *= INV_TEMP; store f32 if n < NCLS                       [GEMM4]
// ---------------------------------------------------------------------------
template <int MODE>
__global__ __launch_bounds__(256)
void gemm_wmma_kernel(const bf16_t* __restrict__ A, const bf16_t* __restrict__ Bw,
                      const float* __restrict__ bias, const float* __restrict__ addend,
                      float* __restrict__ outF, bf16_t* __restrict__ outH,
                      int M, int N, int K, int ldc) {
  __shared__ __align__(16) unsigned char ldsA[2][256 * LDS_ROW_B];  // 2 x 20 KB
  __shared__ __align__(16) unsigned char ldsB[2][128 * LDS_ROW_B];  // 2 x 10 KB

  const int lane  = threadIdx.x & 31;
  const int wave  = threadIdx.x >> 5;
  const int waveM = wave >> 1;                 // 4 waves along M
  const int waveN = wave & 1;                  // 2 waves along N
  const int wgM   = blockIdx.y * 256;
  const int wgN   = blockIdx.x * 128;

  const bf16_t* aTile = A  + (size_t)wgM * K;
  const bf16_t* bTile = Bw + (size_t)wgN * K;

  v8f acc[4][4];
#pragma unroll
  for (int i = 0; i < 4; ++i)
#pragma unroll
    for (int j = 0; j < 4; ++j) acc[i][j] = v8f{};

  const int nk = K / 32;
  if (threadIdx.x < 32) {                      // wave 0 drives the TDM
    tdm_load_tile(lds_offset_of(&ldsA[0][0]), aTile, K, M, 32, 256, K);
    tdm_load_tile(lds_offset_of(&ldsB[0][0]), bTile, K, N, 32, 128, K);
  }

  // Per-lane fragment offsets within the LDS tiles (ISA 7.12.2 layouts):
  // A 16x32: row = lane%16, K chunks at (lane/16)*8 and +16 elems
  // B 32x16: row = lane%16, K chunks at (lane/16)*8 and +16 elems
  const int aRowOff = (waveM * 64 + (lane & 15)) * LDS_ROW_B + ((lane >> 4) << 4);
  const int bRowOff = (waveN * 64 + (lane & 15)) * LDS_ROW_B + ((lane >> 4) << 4);

  for (int t = 0; t < nk; ++t) {
    const int cur = t & 1;
    if (threadIdx.x < 32) {
      if (t + 1 < nk) {
        tdm_load_tile(lds_offset_of(&ldsA[cur ^ 1][0]), aTile + (size_t)(t + 1) * 32,
                      K, M, 32, 256, K);
        tdm_load_tile(lds_offset_of(&ldsB[cur ^ 1][0]), bTile + (size_t)(t + 1) * 32,
                      K, N, 32, 128, K);
        __builtin_amdgcn_s_wait_tensorcnt(2);  // current buffer's 2 DMAs done
      } else {
        __builtin_amdgcn_s_wait_tensorcnt(0);
      }
    }
    __syncthreads();                           // publish LDS tile to all waves

    const unsigned char* aBuf = &ldsA[cur][0];
    const unsigned char* bBuf = &ldsB[cur][0];
    v16bf af[4], bfr[4];
#pragma unroll
    for (int i = 0; i < 4; ++i) {
      const unsigned char* p = aBuf + aRowOff + i * (16 * LDS_ROW_B);
      af[i] = lds_frag(p, p + 32);
    }
#pragma unroll
    for (int j = 0; j < 4; ++j) {
      const unsigned char* p = bBuf + bRowOff + j * (16 * LDS_ROW_B);
      bfr[j] = lds_frag(p, p + 32);
    }
#pragma unroll
    for (int i = 0; i < 4; ++i)
#pragma unroll
      for (int j = 0; j < 4; ++j)
        acc[i][j] = __builtin_amdgcn_wmma_f32_16x16x32_bf16(
            false, af[i], false, bfr[j], (short)0, acc[i][j], false, false);

    __syncthreads();                           // tile consumed; safe to overwrite
  }

  // C/D layout: lane L holds N = L%16, M = vgpr + 8*(L/16)
  const int nCol0 = lane & 15;
  const int mHi   = (lane >> 4) << 3;
  const int mWav  = wgM + waveM * 64;
  const int nWav  = wgN + waveN * 64;
#pragma unroll
  for (int i = 0; i < 4; ++i) {
#pragma unroll
    for (int j = 0; j < 4; ++j) {
      const int n = nWav + j * 16 + nCol0;
#pragma unroll
      for (int v = 0; v < 8; ++v) {
        const int m = mWav + i * 16 + mHi + v;
        float val = acc[i][j][v];
        if (MODE == 0) {
          val += bias[n];
          outF[(size_t)m * ldc + n] = val;
          outH[(size_t)m * NPAD + n] = (bf16_t)val;
        } else if (MODE == 1) {
          if (n < NCLS) outF[(size_t)m * ldc + n] = val;
        } else if (MODE == 2) {
          val += addend[(size_t)m * NPAD + n];
          outF[(size_t)m * ldc + n] = val;
        } else {
          val *= INV_TEMP;
          if (n < NCLS) outF[(size_t)m * ldc + n] = val;
        }
      }
    }
  }
}

// ---------------------------------------------------------------------------
// Conversion / padding kernels
// ---------------------------------------------------------------------------
__global__ void cvt_bf16_kernel(const float* __restrict__ src, bf16_t* __restrict__ dst,
                                size_t n) {
  size_t i = (size_t)blockIdx.x * blockDim.x + threadIdx.x;
  size_t stride = (size_t)gridDim.x * blockDim.x;
  for (; i < n; i += stride) dst[i] = (bf16_t)src[i];
}

__global__ void cvt_pad_rows_kernel(const float* __restrict__ src, bf16_t* __restrict__ dst,
                                    int srcRows, int cols, int rowsPad) {
  size_t i = (size_t)blockIdx.x * blockDim.x + threadIdx.x;
  size_t total = (size_t)rowsPad * cols;
  size_t stride = (size_t)gridDim.x * blockDim.x;
  for (; i < total; i += stride) {
    int r = (int)(i / cols);
    dst[i] = (r < srcRows) ? (bf16_t)src[i] : (bf16_t)0.0f;
  }
}

// dst[n*kPad + k] = src[k*cols + n]  (k < srcRows), else 0.  dst is [cols x kPad]
__global__ void cvt_transpose_pad_kernel(const float* __restrict__ src, bf16_t* __restrict__ dst,
                                         int srcRows, int cols, int kPad) {
  size_t i = (size_t)blockIdx.x * blockDim.x + threadIdx.x;
  size_t total = (size_t)cols * kPad;
  size_t stride = (size_t)gridDim.x * blockDim.x;
  for (; i < total; i += stride) {
    int n = (int)(i / kPad);
    int k = (int)(i % kPad);
    dst[i] = (k < srcRows) ? (bf16_t)src[(size_t)k * cols + n] : (bf16_t)0.0f;
  }
}

// ---------------------------------------------------------------------------
// Row-wise kernels: one wave32 per row
// ---------------------------------------------------------------------------
__global__ __launch_bounds__(256)
void mask_softmax_kernel(const float* __restrict__ out1, bf16_t* __restrict__ sm1) {
  const int lane = threadIdx.x & 31;
  const int row  = blockIdx.x * (blockDim.x >> 5) + (threadIdx.x >> 5);
  const float* r = out1 + (size_t)row * NCLS;

  float bestV = -3.4e38f;
  int   bestI = 0;
  for (int c = lane; c < NCLS; c += 32) {
    float v = r[c];
    if (v > bestV || (v == bestV && c < bestI)) { bestV = v; bestI = c; }
  }
#pragma unroll
  for (int off = 16; off > 0; off >>= 1) {
    float ov = __shfl_xor(bestV, off, 32);
    int   oi = __shfl_xor(bestI, off, 32);
    if (ov > bestV || (ov == bestV && oi < bestI)) { bestV = ov; bestI = oi; }
  }
  const int   predict = bestI;
  const float maxv    = bestV;            // masked row-max == value at predict
  const bool  privRow = predict >= SHARE_N;

  float sum = 0.0f;
  for (int c = lane; c < NCLS; c += 32) {
    bool mask = privRow ? true : (c < SHARE_N);
    mask = mask && (c != predict);
    float v = mask ? NEG_FILL : r[c];
    sum += __expf(v - maxv);
  }
#pragma unroll
  for (int off = 16; off > 0; off >>= 1) sum += __shfl_xor(sum, off, 32);
  const float inv = 1.0f / sum;

  bf16_t* o = sm1 + (size_t)row * NPAD;
  for (int c = lane; c < NPAD; c += 32) {
    float p = 0.0f;
    if (c < NCLS) {
      bool mask = privRow ? true : (c < SHARE_N);
      mask = mask && (c != predict);
      float v = mask ? NEG_FILL : r[c];
      p = __expf(v - maxv) * inv;
    }
    o[c] = (bf16_t)p;
  }
}

__global__ __launch_bounds__(256)
void normalize_kernel(const float* __restrict__ faug, bf16_t* __restrict__ xout) {
  const int lane = threadIdx.x & 31;
  const int row  = blockIdx.x * (blockDim.x >> 5) + (threadIdx.x >> 5);
  const float* r = faug + (size_t)row * D_BOT;
  float ss = 0.0f;
  for (int c = lane; c < D_BOT; c += 32) { float v = r[c]; ss += v * v; }
#pragma unroll
  for (int off = 16; off > 0; off >>= 1) ss += __shfl_xor(ss, off, 32);
  const float inv = 1.0f / fmaxf(sqrtf(ss), NORM_EPS);
  bf16_t* o = xout + (size_t)row * D_BOT;
  for (int c = lane; c < D_BOT; c += 32) o[c] = (bf16_t)(r[c] * inv);
}

__global__ __launch_bounds__(256)
void softmax_kernel(const float* __restrict__ out2, float* __restrict__ sm) {
  const int lane = threadIdx.x & 31;
  const int row  = blockIdx.x * (blockDim.x >> 5) + (threadIdx.x >> 5);
  const float* r = out2 + (size_t)row * NCLS;
  float mx = -3.4e38f;
  for (int c = lane; c < NCLS; c += 32) mx = fmaxf(mx, r[c]);
#pragma unroll
  for (int off = 16; off > 0; off >>= 1) mx = fmaxf(mx, __shfl_xor(mx, off, 32));
  float sum = 0.0f;
  for (int c = lane; c < NCLS; c += 32) sum += __expf(r[c] - mx);
#pragma unroll
  for (int off = 16; off > 0; off >>= 1) sum += __shfl_xor(sum, off, 32);
  const float inv = 1.0f / sum;
  float* o = sm + (size_t)row * NCLS;
  for (int c = lane; c < NCLS; c += 32) o[c] = __expf(r[c] - mx) * inv;
}

// ---------------------------------------------------------------------------
extern "C" void kernel_launch(void* const* d_in, const int* in_sizes, int n_in,
                              void* d_out, int out_size, void* d_ws, size_t ws_size,
                              hipStream_t stream) {
  (void)in_sizes; (void)n_in; (void)out_size; (void)ws_size;

  const float* features = (const float*)d_in[0];   // [8192, 2048]
  const float* W_b      = (const float*)d_in[1];   // [1024, 2048]
  const float* b_b      = (const float*)d_in[2];   // [1024]
  const float* W1       = (const float*)d_in[3];   // [1000, 1024]
  const float* W2       = (const float*)d_in[4];   // [1000, 1024]
  const float* centroid = (const float*)d_in[5];   // [1000, 1024]

  float* out       = (float*)d_out;
  float* feats_out = out;                                      // [8192,1024]
  float* faug_out  = feats_out + (size_t)B_ROWS * D_BOT;       // [8192,1024]
  float* out1      = faug_out  + (size_t)B_ROWS * D_BOT;       // [8192,1000]
  float* out2      = out1      + (size_t)B_ROWS * NCLS;        // [8192,1000]
  float* smout     = out2      + (size_t)B_ROWS * NCLS;        // [8192,1000]

  char* ws = (char*)d_ws;
  bf16_t* feat_bf = (bf16_t*)ws; ws += (size_t)B_ROWS * D_IN  * sizeof(bf16_t);
  bf16_t* wb_bf   = (bf16_t*)ws; ws += (size_t)D_BOT  * D_IN  * sizeof(bf16_t);
  bf16_t* w1_bf   = (bf16_t*)ws; ws += (size_t)NPAD   * D_BOT * sizeof(bf16_t);
  bf16_t* w2_bf   = (bf16_t*)ws; ws += (size_t)NPAD   * D_BOT * sizeof(bf16_t);
  bf16_t* centT   = (bf16_t*)ws; ws += (size_t)D_BOT  * NPAD  * sizeof(bf16_t);
  bf16_t* bot_bf  = (bf16_t*)ws; ws += (size_t)B_ROWS * NPAD  * sizeof(bf16_t);
  bf16_t* sm1_bf  = (bf16_t*)ws; ws += (size_t)B_ROWS * NPAD  * sizeof(bf16_t);
  bf16_t* x_bf    = (bf16_t*)ws; ws += (size_t)B_ROWS * NPAD  * sizeof(bf16_t);

  // --- operand conversion to bf16 (all GEMM operands stay L2-resident) ---
  cvt_bf16_kernel<<<2048, 256, 0, stream>>>(features, feat_bf, (size_t)B_ROWS * D_IN);
  cvt_bf16_kernel<<<1024, 256, 0, stream>>>(W_b, wb_bf, (size_t)D_BOT * D_IN);
  cvt_pad_rows_kernel<<<512, 256, 0, stream>>>(W1, w1_bf, NCLS, D_BOT, NPAD);
  cvt_pad_rows_kernel<<<512, 256, 0, stream>>>(W2, w2_bf, NCLS, D_BOT, NPAD);
  cvt_transpose_pad_kernel<<<512, 256, 0, stream>>>(centroid, centT, NCLS, D_BOT, NPAD);

  const dim3 blk(256);
  const dim3 gBot(D_BOT / 128, B_ROWS / 256);   // 8 x 32
  const dim3 gCls(NPAD  / 128, B_ROWS / 256);   // 8 x 32

  // GEMM1: feats = features @ W_b^T + b_b   (f32 + bf16 dual store)
  gemm_wmma_kernel<0><<<gBot, blk, 0, stream>>>(feat_bf, wb_bf, b_b, nullptr,
                                                feats_out, bot_bf,
                                                B_ROWS, D_BOT, D_IN, D_BOT);
  // GEMM2: outputs1 = feats @ W1^T
  gemm_wmma_kernel<1><<<gCls, blk, 0, stream>>>(bot_bf, w1_bf, nullptr, nullptr,
                                                out1, nullptr,
                                                B_ROWS, NPAD, D_BOT, NCLS);
  // masked softmax -> sm1 (bf16, cols 1000..1023 zero)
  mask_softmax_kernel<<<B_ROWS / 8, 256, 0, stream>>>(out1, sm1_bf);
  // GEMM3: features_aug = sm1 @ centroid + feats
  gemm_wmma_kernel<2><<<gBot, blk, 0, stream>>>(sm1_bf, centT, nullptr, feats_out,
                                                faug_out, nullptr,
                                                B_ROWS, D_BOT, NPAD, D_BOT);
  // row L2-normalize -> x (bf16)
  normalize_kernel<<<B_ROWS / 8, 256, 0, stream>>>(faug_out, x_bf);
  // GEMM4: outputs2 = x @ W2^T / temp
  gemm_wmma_kernel<3><<<gCls, blk, 0, stream>>>(x_bf, w2_bf, nullptr, nullptr,
                                                out2, nullptr,
                                                B_ROWS, NPAD, D_BOT, NCLS);
  // final softmax
  softmax_kernel<<<B_ROWS / 8, 256, 0, stream>>>(out2, smout);
}
